// Attention_77060303225320
// MI455X (gfx1250) — matrix operand order repaired and checked by reading, not compiled, not run on hardware
//
#include <hip/hip_runtime.h>
#include <hip/hip_bf16.h>
#include <math.h>

// ---------------------------------------------------------------------------
// MI455X (gfx1250) attention with RoPE.
// bf16 WMMA (v_wmma_f32_16x16x32_bf16) for all GEMMs + flash attention.
// Async global->LDS staging (ASYNCcnt) with LDS double buffering when the
// toolchain exposes the gfx1250 async builtins.
// ---------------------------------------------------------------------------

typedef __attribute__((ext_vector_type(16))) __bf16 v16bf;
typedef __attribute__((ext_vector_type(8)))  float  v8f;
typedef __attribute__((ext_vector_type(4)))  __bf16 v4bf;
typedef __attribute__((ext_vector_type(4)))  float  f4_t;

#define DMODEL 1024
#define SEQ    1024
#define BATCH  16
#define NHEADS 16
#define HDIM   64
#define MROWS  (BATCH * SEQ)   // 16384

#if __has_builtin(__builtin_amdgcn_global_load_async_to_lds_b128) && \
    __has_builtin(__builtin_amdgcn_s_wait_asynccnt)
#define HAVE_ASYNC_LDS 1
#else
#define HAVE_ASYNC_LDS 0
#endif

// Pointer types for the async-to-LDS builtin:
//   (v4i addrspace(1)*, v4i addrspace(3)*, imm offset, imm cpol)
typedef int v4i_t __attribute__((vector_size(16)));
typedef __attribute__((address_space(1))) v4i_t g1_v4i;
typedef __attribute__((address_space(3))) v4i_t l3_v4i;

// Fragment container: 16 bf16 = 32 bytes = 8 VGPRs (one WMMA A or B operand).
union Frag {
    v16bf v;
    f4_t  f4[2];
};

// Copy 32 bytes (16 bf16) global -> LDS. Async (ASYNCcnt) when available.
__device__ __forceinline__ void cp32_g2l(const __bf16* g, __bf16* l)
{
#if HAVE_ASYNC_LDS
    g1_v4i* gp = (g1_v4i*)g;
    l3_v4i* lp = (l3_v4i*)l;
    __builtin_amdgcn_global_load_async_to_lds_b128(gp,     lp,     0, 0);
    __builtin_amdgcn_global_load_async_to_lds_b128(gp + 1, lp + 1, 0, 0);
#else
    ((f4_t*)l)[0] = ((const f4_t*)g)[0];
    ((f4_t*)l)[1] = ((const f4_t*)g)[1];
#endif
}

__device__ __forceinline__ void wait_g2l()
{
#if HAVE_ASYNC_LDS
    __builtin_amdgcn_s_wait_asynccnt(0);
#endif
}

// ---------------------------------------------------------------------------
// f32 -> bf16 conversion (vectorized 4-wide)
// ---------------------------------------------------------------------------
__global__ __launch_bounds__(256)
void cvt_f32_bf16(const float* __restrict__ in, __bf16* __restrict__ out, int n4)
{
    int i = blockIdx.x * 256 + threadIdx.x;
    if (i >= n4) return;
    f4_t v = ((const f4_t*)in)[i];
    ((v4bf*)out)[i] = __builtin_convertvector(v, v4bf);
}

// ---------------------------------------------------------------------------
// GEMM: out[M,N] = A[M,K](bf16) @ W[N,K]^T(bf16) + bias (f32 accumulate)
// Block tile 128x128, K-step 32, double-buffered LDS, async staging.
// 8 waves: 4 (M) x 2 (N); each wave 32x64 = 2x4 WMMA accumulators.
// A-frag / B-frag both use contiguous k = 16*(lane/16) + s (16-bit 16x32 A, 32x16 B).
// ---------------------------------------------------------------------------
template<bool OUT_BF16>
__global__ __launch_bounds__(256)
void gemm_bf16(const __bf16* __restrict__ A, const __bf16* __restrict__ W,
               const float* __restrict__ bias, void* __restrict__ out,
               int M, int N, int K)
{
    __shared__ __bf16 As[2][128 * 32];   // [buf][row][k] 2 x 8 KB
    __shared__ __bf16 Bs[2][128 * 32];   // [buf][col][k] 2 x 8 KB

    const int tid  = threadIdx.x;
    const int lane = tid & 31;
    const int wid  = tid >> 5;
    const int m0   = blockIdx.y * 128;
    const int n0   = blockIdx.x * 128;
    const int wm   = (wid & 3) * 32;   // wave M base inside block
    const int wn   = (wid >> 2) * 64;  // wave N base inside block
    const int crow = tid >> 1;         // 0..127 cooperative load row
    const int cseg = (tid & 1) * 16;   // 0 or 16 (bf16 elements)

    const __bf16* gA = A + (size_t)(m0 + crow) * K + cseg;
    const __bf16* gW = W + (size_t)(n0 + crow) * K + cseg;
    __bf16* lA0 = &As[0][crow * 32 + cseg];
    __bf16* lB0 = &Bs[0][crow * 32 + cseg];
    __bf16* lA1 = &As[1][crow * 32 + cseg];
    __bf16* lB1 = &Bs[1][crow * 32 + cseg];

    v8f acc[2][4] = {};

    const int khalf = (lane >> 4) * 8;   // unused (A-frag loads use bk0)
    const int bk0   = (lane >> 4) * 16;  // B-frag K base
    const int ln15  = lane & 15;

    // prologue: stage tile 0 into buffer 0
    cp32_g2l(gA, lA0);
    cp32_g2l(gW, lB0);

    for (int kk = 0, buf = 0; kk < K; kk += 32, buf ^= 1) {
        wait_g2l();
        __syncthreads();

        // stage next tile into the other buffer while we compute
        if (kk + 32 < K) {
            cp32_g2l(gA + kk + 32, buf ? lA0 : lA1);
            cp32_g2l(gW + kk + 32, buf ? lB0 : lB1);
        }

        const __bf16* as = As[buf];
        const __bf16* bs = Bs[buf];
        Frag af[2], bf[4];
#pragma unroll
        for (int im = 0; im < 2; ++im) {
            const __bf16* p = as + (wm + im * 16 + ln15) * 32;
            af[im].f4[0] = *(const f4_t*)(p + bk0);
            af[im].f4[1] = *(const f4_t*)(p + bk0 + 8);
        }
#pragma unroll
        for (int jn = 0; jn < 4; ++jn) {
            const __bf16* p = bs + (wn + jn * 16 + ln15) * 32 + bk0;
            bf[jn].f4[0] = ((const f4_t*)p)[0];
            bf[jn].f4[1] = ((const f4_t*)p)[1];
        }
#pragma unroll
        for (int im = 0; im < 2; ++im)
#pragma unroll
            for (int jn = 0; jn < 4; ++jn)
                acc[im][jn] = __builtin_amdgcn_wmma_f32_16x16x32_bf16(
                    false, af[im].v, false, bf[jn].v,
                    (short)0, acc[im][jn], false, false);
        // no trailing barrier: writes for iteration i+2 target the buffer
        // last read at iteration i, and every wave passes the top-of-loop
        // barrier of iteration i+1 after that read.
    }

    // epilogue: bias add + store (C-layout: VGPR i -> M = i + 8*(lane>=16))
#pragma unroll
    for (int im = 0; im < 2; ++im) {
#pragma unroll
        for (int jn = 0; jn < 4; ++jn) {
            const int colg = n0 + wn + jn * 16 + ln15;
            const float bv = bias[colg];
#pragma unroll
            for (int i = 0; i < 8; ++i) {
                const int rowg = m0 + wm + im * 16 + (lane >> 4) * 8 + i;
                const float v = acc[im][jn][i] + bv;
                if (OUT_BF16) ((__bf16*)out)[(size_t)rowg * N + colg] = (__bf16)v;
                else          ((float*)out)[(size_t)rowg * N + colg] = v;
            }
        }
    }
}

// ---------------------------------------------------------------------------
// RoPE: 1D on dims [0,16), 2D on dims [16,64) of each head; optional scale
// (q gets 1/sqrt(HDIM) folded in here). In-place on bf16 tensor [MROWS,DMODEL].
// ---------------------------------------------------------------------------
__global__ __launch_bounds__(256)
void rope_kernel(__bf16* __restrict__ T, const int* __restrict__ widthp, float scale)
{
    const int gid  = blockIdx.x * 256 + threadIdx.x;  // over MROWS*NHEADS
    const int row  = gid >> 4;
    const int head = gid & 15;
    if (row >= MROWS) return;
    const int s     = row & (SEQ - 1);
    const int width = widthp[0];
    __bf16* p = T + (size_t)row * DMODEL + head * HDIM;
    const float lb = logf(10000.0f);

    // 1D RoPE on latent dims: pairs (2j, 2j+1), inv_freq = base^(-2j/16)
#pragma unroll
    for (int j = 0; j < 8; ++j) {
        const float inv = __expf(-(2.0f * j / 16.0f) * lb);
        const float ang = (float)s * inv;
        const float c = __cosf(ang), sn = __sinf(ang);
        const float a = (float)p[2 * j], b = (float)p[2 * j + 1];
        p[2 * j]     = (__bf16)(scale * (a * c - b * sn));
        p[2 * j + 1] = (__bf16)(scale * (a * sn + b * c));
    }
    // 2D RoPE on spatial dims: quarters of 12, inv_freq = base^(-4j/48)
    const int px = s % width;
    const int py = s / width;
#pragma unroll
    for (int j = 0; j < 12; ++j) {
        const float inv = __expf(-(4.0f * j / 48.0f) * lb);
        const float ax = (float)px * inv, ay = (float)py * inv;
        const float cx = __cosf(ax), sx = __sinf(ax);
        const float cy = __cosf(ay), sy = __sinf(ay);
        const float xr = (float)p[16 + j], xi = (float)p[28 + j];
        const float yr = (float)p[40 + j], yi = (float)p[52 + j];
        p[16 + j] = (__bf16)(scale * (xr * cx - xi * sx));
        p[28 + j] = (__bf16)(scale * (xr * sx + xi * cx));
        p[40 + j] = (__bf16)(scale * (yr * cy - yi * sy));
        p[52 + j] = (__bf16)(scale * (yr * sy + yi * cy));
    }
}

// ---------------------------------------------------------------------------
// Flash attention. One block = (batch, head, 128-query tile); 8 waves, each
// wave owns 16 query rows. KV processed in 64-row blocks staged in LDS
// (K via async-to-LDS, V via register transpose). Scores and P@V via bf16
// WMMA; P goes through per-wave LDS for C->A layout conversion.
// Scale 1/sqrt(64) is pre-folded into q by rope_kernel.
// ---------------------------------------------------------------------------
__global__ __launch_bounds__(256)
void attn_kernel(const __bf16* __restrict__ Q, const __bf16* __restrict__ Km,
                 const __bf16* __restrict__ Vm, __bf16* __restrict__ O)
{
    __shared__ __bf16 Ks[64 * 64];       // [t][d] 8 KB
    __shared__ __bf16 Vs[64 * 64];       // transposed [d][t] 8 KB
    __shared__ __bf16 Ps[8 * 16 * 64];   // per-wave P tile 16 KB

    const int tid  = threadIdx.x;
    const int lane = tid & 31;
    const int wid  = tid >> 5;
    const int bid  = blockIdx.x;
    const int qt   = bid & 7;            // SEQ/128 = 8 query tiles
    const int head = (bid >> 3) & 15;
    const int b    = bid >> 7;

    const size_t rowBase = (size_t)b * SEQ;
    const int    colBase = head * HDIM;
    const int    ln15    = lane & 15;
    const int    khalf   = (lane >> 4) * 8;
    const int    bk0     = (lane >> 4) * 16;

    // Q fragments (16 rows x 64 dims -> 2 A-frags over K=32 chunks)
    const int qrow = qt * 128 + wid * 16 + ln15;
    Frag qf[2];
#pragma unroll
    for (int kc = 0; kc < 2; ++kc) {
        const __bf16* p = Q + (rowBase + qrow) * DMODEL + colBase + kc * 32;
        qf[kc].f4[0] = *(const f4_t*)(p + bk0);
        qf[kc].f4[1] = *(const f4_t*)(p + bk0 + 8);
    }

    float mrow[8], lrow[8];
    v8f oacc[4] = {};
#pragma unroll
    for (int i = 0; i < 8; ++i) { mrow[i] = -1e30f; lrow[i] = 0.0f; }

    const int cr = tid >> 2;         // 0..63 cooperative load row
    const int cs = (tid & 3) * 16;   // 0,16,32,48

    for (int t0 = 0; t0 < SEQ; t0 += 64) {
        // stage K (async, row-major) and V (register transpose) into LDS
        {
            cp32_g2l(Km + (rowBase + t0 + cr) * DMODEL + colBase + cs,
                     Ks + cr * 64 + cs);
            Frag vv;
            const f4_t* pv = (const f4_t*)(Vm + (rowBase + t0 + cr) * DMODEL + colBase + cs);
            vv.f4[0] = pv[0]; vv.f4[1] = pv[1];
#pragma unroll
            for (int e = 0; e < 16; ++e)
                Vs[(cs + e) * 64 + cr] = vv.v[e];
            if (t0 + 64 < SEQ) {   // global_prefetch_b8 for next KV block
                __builtin_prefetch(Km + (rowBase + t0 + 64 + cr) * DMODEL + colBase + cs, 0, 1);
                __builtin_prefetch(Vm + (rowBase + t0 + 64 + cr) * DMODEL + colBase + cs, 0, 1);
            }
        }
        wait_g2l();
        __syncthreads();

        // scores: S(16x64) = Q(16x64) @ K^T  -> 4 column tiles, 2 WMMA each
        v8f sc[4];
#pragma unroll
        for (int nt = 0; nt < 4; ++nt) {
            Frag kf0, kf1;
            const __bf16* p = Ks + (nt * 16 + ln15) * 64;
            kf0.f4[0] = ((const f4_t*)(p + bk0))[0];
            kf0.f4[1] = ((const f4_t*)(p + bk0))[1];
            kf1.f4[0] = ((const f4_t*)(p + 32 + bk0))[0];
            kf1.f4[1] = ((const f4_t*)(p + 32 + bk0))[1];
            v8f s = {};
            s = __builtin_amdgcn_wmma_f32_16x16x32_bf16(false, qf[0].v, false, kf0.v,
                                                        (short)0, s, false, false);
            s = __builtin_amdgcn_wmma_f32_16x16x32_bf16(false, qf[1].v, false, kf1.v,
                                                        (short)0, s, false, false);
            sc[nt] = s;
        }

        // online softmax: row max across 16-lane group + 4 tiles
        float mnew[8];
#pragma unroll
        for (int i = 0; i < 8; ++i) mnew[i] = mrow[i];
#pragma unroll
        for (int nt = 0; nt < 4; ++nt)
#pragma unroll
            for (int i = 0; i < 8; ++i) mnew[i] = fmaxf(mnew[i], sc[nt][i]);
#pragma unroll
        for (int off = 1; off < 16; off <<= 1)
#pragma unroll
            for (int i = 0; i < 8; ++i)
                mnew[i] = fmaxf(mnew[i], __shfl_xor(mnew[i], off, 32));

        float corr[8];
#pragma unroll
        for (int i = 0; i < 8; ++i) {
            corr[i] = __expf(mrow[i] - mnew[i]);
            lrow[i] *= corr[i];
            mrow[i] = mnew[i];
        }
#pragma unroll
        for (int nt = 0; nt < 4; ++nt) {
#pragma unroll
            for (int i = 0; i < 8; ++i) {
                const float pv = __expf(sc[nt][i] - mnew[i]);
                lrow[i] += pv;                 // lane-partial row sum
                oacc[nt][i] *= corr[i];
                Ps[wid * 1024 + (i + 8 * (lane >> 4)) * 64 + nt * 16 + ln15] = (__bf16)pv;
            }
        }

        // P (C-layout in LDS) reloaded as A-frags; O += P @ V
        Frag pf[2];
#pragma unroll
        for (int kc = 0; kc < 2; ++kc) {
            const __bf16* p = Ps + wid * 1024 + ln15 * 64 + kc * 32;
            pf[kc].f4[0] = *(const f4_t*)(p + bk0);
            pf[kc].f4[1] = *(const f4_t*)(p + bk0 + 8);
        }
#pragma unroll
        for (int nt = 0; nt < 4; ++nt) {
            Frag vf0, vf1;
            const __bf16* p = Vs + (nt * 16 + ln15) * 64;   // Vs[d][t]
            vf0.f4[0] = ((const f4_t*)(p + bk0))[0];
            vf0.f4[1] = ((const f4_t*)(p + bk0))[1];
            vf1.f4[0] = ((const f4_t*)(p + 32 + bk0))[0];
            vf1.f4[1] = ((const f4_t*)(p + 32 + bk0))[1];
            oacc[nt] = __builtin_amdgcn_wmma_f32_16x16x32_bf16(false, pf[0].v, false, vf0.v,
                                                               (short)0, oacc[nt], false, false);
            oacc[nt] = __builtin_amdgcn_wmma_f32_16x16x32_bf16(false, pf[1].v, false, vf1.v,
                                                               (short)0, oacc[nt], false, false);
        }
        __syncthreads();
    }

    // finalize: reduce row sums across lane group, normalize, store bf16
#pragma unroll
    for (int off = 1; off < 16; off <<= 1)
#pragma unroll
        for (int i = 0; i < 8; ++i) lrow[i] += __shfl_xor(lrow[i], off, 32);

    float invl[8];
#pragma unroll
    for (int i = 0; i < 8; ++i) invl[i] = 1.0f / lrow[i];

#pragma unroll
    for (int nt = 0; nt < 4; ++nt) {
        const int col = colBase + nt * 16 + ln15;
#pragma unroll
        for (int i = 0; i < 8; ++i) {
            const int row = qt * 128 + wid * 16 + i + 8 * (lane >> 4);
            O[(rowBase + row) * DMODEL + col] = (__bf16)(oacc[nt][i] * invl[i]);
        }
    }
}

// ---------------------------------------------------------------------------
// Host launcher
// ---------------------------------------------------------------------------
extern "C" void kernel_launch(void* const* d_in, const int* in_sizes, int n_in,
                              void* d_out, int out_size, void* d_ws, size_t ws_size,
                              hipStream_t stream)
{
    (void)in_sizes; (void)n_in; (void)out_size; (void)ws_size;
    const float* x  = (const float*)d_in[0];
    const float* Wq = (const float*)d_in[1];
    const float* bq = (const float*)d_in[2];
    const float* Wk = (const float*)d_in[3];
    const float* bk = (const float*)d_in[4];
    const float* Wv = (const float*)d_in[5];
    const float* bv = (const float*)d_in[6];
    const float* Wo = (const float*)d_in[7];
    const float* bo = (const float*)d_in[8];
    const int* width = (const int*)d_in[10];
    float* out = (float*)d_out;

    // workspace layout (bf16 elements), ~176 MB total
    __bf16* ws  = (__bf16*)d_ws;
    __bf16* xb  = ws;
    __bf16* wqb = xb  + (size_t)MROWS * DMODEL;
    __bf16* wkb = wqb + (size_t)DMODEL * DMODEL;
    __bf16* wvb = wkb + (size_t)DMODEL * DMODEL;
    __bf16* wob = wvb + (size_t)DMODEL * DMODEL;
    __bf16* qb  = wob + (size_t)DMODEL * DMODEL;
    __bf16* kb  = qb  + (size_t)MROWS * DMODEL;
    __bf16* vb  = kb  + (size_t)MROWS * DMODEL;
    __bf16* ob  = vb  + (size_t)MROWS * DMODEL;

    const int nX4 = MROWS * DMODEL / 4;
    const int nW4 = DMODEL * DMODEL / 4;
    cvt_f32_bf16<<<nX4 / 256, 256, 0, stream>>>(x,  xb,  nX4);
    cvt_f32_bf16<<<nW4 / 256, 256, 0, stream>>>(Wq, wqb, nW4);
    cvt_f32_bf16<<<nW4 / 256, 256, 0, stream>>>(Wk, wkb, nW4);
    cvt_f32_bf16<<<nW4 / 256, 256, 0, stream>>>(Wv, wvb, nW4);
    cvt_f32_bf16<<<nW4 / 256, 256, 0, stream>>>(Wo, wob, nW4);

    dim3 gg(DMODEL / 128, MROWS / 128);
    gemm_bf16<true><<<gg, 256, 0, stream>>>(xb, wqb, bq, qb, MROWS, DMODEL, DMODEL);
    gemm_bf16<true><<<gg, 256, 0, stream>>>(xb, wkb, bk, kb, MROWS, DMODEL, DMODEL);
    gemm_bf16<true><<<gg, 256, 0, stream>>>(xb, wvb, bv, vb, MROWS, DMODEL, DMODEL);

    const int nR = MROWS * NHEADS / 256;
    rope_kernel<<<nR, 256, 0, stream>>>(qb, width, 0.125f);  // 1/sqrt(HDIM) folded in
    rope_kernel<<<nR, 256, 0, stream>>>(kb, width, 1.0f);

    attn_kernel<<<BATCH * NHEADS * (SEQ / 128), 256, 0, stream>>>(qb, kb, vb, ob);

    gemm_bf16<false><<<gg, 256, 0, stream>>>(ob, wob, bo, out, MROWS, DMODEL, DMODEL);
}
